// MultiHeadSelfAttention_88003879895091
// MI455X (gfx1250) — matrix operand order repaired and checked by reading, not compiled, not run on hardware
//
#include <hip/hip_runtime.h>
#include <hip/hip_bf16.h>

// ---------------------------------------------------------------------------
// Multi-head self-attention for MI455X (gfx1250, wave32, WMMA).
//   B=4, S=2048, D=1024, H=16, PD=64.  Compute-bound (~140 GFLOP vs ~160 MB):
//   all matmuls via v_wmma_f32_16x16x32_f16; flash-attention (online softmax).
//   Tile staging uses CDNA5 GLOBAL_LOAD_ASYNC_TO_LDS_B128 (ASYNCcnt) with
//   double-buffered LDS: one barrier per K-step, async engine overlaps WMMA.
// ---------------------------------------------------------------------------

typedef __attribute__((ext_vector_type(16))) _Float16 v16h;
typedef __attribute__((ext_vector_type(4)))  _Float16 v4h;
typedef __attribute__((ext_vector_type(8)))  float    v8f;

constexpr int Bc  = 4;
constexpr int Sc  = 2048;
constexpr int Dc  = 1024;
constexpr int Hc  = 16;
constexpr int PDc = 64;
constexpr int Mtot = Bc * Sc;          // 8192 rows for the projections

#define MODE_QK  0   // store f16 at [b,h,s,pd]
#define MODE_VT  1   // store f16 at [b,h,pd,s]  (V transposed for P@V)
#define MODE_F32 2   // store f32 row-major (final output)

// Build a 16-half fragment from two 16-byte LDS chunks.
__device__ inline v16h ldfrag(const _Float16* p0, const _Float16* p1) {
  v16h r;
  reinterpret_cast<float4*>(&r)[0] = *reinterpret_cast<const float4*>(p0);
  reinterpret_cast<float4*>(&r)[1] = *reinterpret_cast<const float4*>(p1);
  return r;
}

// CDNA5 async memory->LDS copy, 16 bytes per lane.  Tracked with ASYNCcnt.
// LDS address = low 32 bits of the generic pointer (ISA 10.2 aperture rule).
__device__ inline void async_copy_b128(void* lds, const void* gptr) {
  unsigned l = (unsigned)(size_t)lds;
  unsigned long long a = (unsigned long long)gptr;
  asm volatile("global_load_async_to_lds_b128 %0, %1, off"
               :: "v"(l), "v"(a) : "memory");
}
__device__ inline void wait_async0() {
  asm volatile("s_wait_asynccnt 0x0" ::: "memory");
}

// ---------------------------------------------------------------------------
// f32 -> f16 conversion (vectorized 4-wide)
// ---------------------------------------------------------------------------
__global__ void cvt_f32_f16(const float* __restrict__ in,
                            _Float16* __restrict__ out, int n4) {
  int i = blockIdx.x * blockDim.x + threadIdx.x;
  if (i < n4) {
    float4 v = reinterpret_cast<const float4*>(in)[i];
    v4h h;
    h.x = (_Float16)v.x; h.y = (_Float16)v.y;
    h.z = (_Float16)v.z; h.w = (_Float16)v.w;
    reinterpret_cast<v4h*>(out)[i] = h;
  }
}

// W[k][n] f32  ->  Wt[n][k] f16  (B-matrix wants columns contiguous over K)
__global__ void cvt_transpose(const float* __restrict__ W,
                              _Float16* __restrict__ Wt) {
  int idx = blockIdx.x * blockDim.x + threadIdx.x;   // Dc*Dc threads
  int k = idx >> 10, n = idx & (Dc - 1);
  Wt[(size_t)n * Dc + k] = (_Float16)W[idx];
}

// ---------------------------------------------------------------------------
// Tiled WMMA GEMM: C[M,N] = A[M,K] * Bt[N,K]^T + bias
//   Block tile 128x128, K-tile 64 (two wmma k-steps), double-buffered LDS
//   filled by async-to-LDS.  8 waves: 2 (M) x 4 (N); each wave 64x32.
// ---------------------------------------------------------------------------
template <int MODE>
__global__ __launch_bounds__(256)
void gemm_wmma(const _Float16* __restrict__ A,
               const _Float16* __restrict__ Bt,
               const float*    __restrict__ bias,
               void* __restrict__ outp) {
  constexpr int BM = 128, BN = 128, BK = 64, LDT = BK + 8;
  __shared__ _Float16 As[2][BM][LDT];
  __shared__ _Float16 Bs[2][BN][LDT];

  const int tid  = threadIdx.x;
  const int lane = tid & 31;
  const int wave = tid >> 5;
  const int l16  = lane & 15;
  const int g    = lane >> 4;          // half-wave selector
  const int waveN = wave & 3;          // 4 waves over N (4*32 = 128)
  const int waveM = wave >> 2;         // 2 waves over M (2*64 = 128)
  const int mBase = blockIdx.y * BM;
  const int nBase = blockIdx.x * BN;

  // this thread's 4 staging chunks (8 halves each); 1024 chunks per tile
  auto issue_tiles = [&](int buf, int kb) {
#pragma unroll
    for (int it = 0; it < 4; ++it) {
      int chunk = tid + it * 256;
      int r = chunk >> 3, c = (chunk & 7) * 8;
      async_copy_b128(&As[buf][r][c], A  + (size_t)(mBase + r) * Dc + kb + c);
      async_copy_b128(&Bs[buf][r][c], Bt + (size_t)(nBase + r) * Dc + kb + c);
    }
  };

  v8f acc[4][2];
  v8f zero = {};
#pragma unroll
  for (int mi = 0; mi < 4; ++mi)
#pragma unroll
    for (int ni = 0; ni < 2; ++ni) acc[mi][ni] = zero;

  issue_tiles(0, 0);

  for (int kb = 0; kb < Dc; kb += BK) {
    const int buf = (kb / BK) & 1;
    wait_async0();          // my async fills for `buf` are in LDS
    __syncthreads();        // everyone's are; readers of buf^1 are done
    if (kb + BK < Dc) issue_tiles(buf ^ 1, kb + BK);

#pragma unroll
    for (int kk = 0; kk < BK; kk += 32) {
      v16h a[4], b[2];
#pragma unroll
      for (int mi = 0; mi < 4; ++mi) {
        // A layout: lane group g -> K g*8..g*8+7 and +16 (ISA 16-bit A 16x32)
        const _Float16* pr = &As[buf][waveM * 64 + mi * 16 + l16][kk + g * 8];
        a[mi] = ldfrag(pr, pr + 16);
      }
#pragma unroll
      for (int ni = 0; ni < 2; ++ni) {
        // B layout: lane group g -> K g*8..g*8+7 and +16 (matches A)
        const _Float16* pr = &Bs[buf][waveN * 32 + ni * 16 + l16][kk + g * 8];
        b[ni] = ldfrag(pr, pr + 16);
      }
#pragma unroll
      for (int mi = 0; mi < 4; ++mi)
#pragma unroll
        for (int ni = 0; ni < 2; ++ni)
          acc[mi][ni] = __builtin_amdgcn_wmma_f32_16x16x32_f16(
              false, a[mi], false, b[ni], (short)0, acc[mi][ni], false, false);
    }
  }

  // Epilogue.  C layout: lane = column (l16), VGPR i = row i + 8*g.
#pragma unroll
  for (int mi = 0; mi < 4; ++mi) {
    int gm0 = mBase + waveM * 64 + mi * 16 + 8 * g;
#pragma unroll
    for (int ni = 0; ni < 2; ++ni) {
      int gn = nBase + waveN * 32 + ni * 16 + l16;
      float bv = bias[gn];
      const float* ap = reinterpret_cast<const float*>(&acc[mi][ni]);
#pragma unroll
      for (int i = 0; i < 8; ++i) {
        int gm = gm0 + i;
        float v = ap[i] + bv;
        if constexpr (MODE == MODE_F32) {
          reinterpret_cast<float*>(outp)[(size_t)gm * Dc + gn] = v;
        } else {
          int b = gm >> 11, s = gm & (Sc - 1);     // S = 2048
          int h = gn >> 6,  pd = gn & (PDc - 1);   // PD = 64
          _Float16 hv = (_Float16)v;
          if constexpr (MODE == MODE_QK)
            reinterpret_cast<_Float16*>(outp)
                [(((size_t)(b * Hc + h)) * Sc + s) * PDc + pd] = hv;
          else
            reinterpret_cast<_Float16*>(outp)
                [(((size_t)(b * Hc + h)) * PDc + pd) * Sc + s] = hv;
        }
      }
    }
  }
}

// ---------------------------------------------------------------------------
// Flash attention: grid = (S/128, B*H), 256 threads.
//   Wave w owns 16 query rows; Q fragments in registers; streams 64-key
//   tiles of K [s,pd] and Vt [pd,s] through double-buffered LDS (async fill)
//   with online softmax.
// ---------------------------------------------------------------------------
__global__ __launch_bounds__(256)
void flash_attn(const _Float16* __restrict__ Qh,
                const _Float16* __restrict__ Kh,
                const _Float16* __restrict__ Vt,
                _Float16* __restrict__ attn) {
  constexpr int KB = 64, LDK = 72;
  __shared__ _Float16 Ks[2][KB][LDK];     // [key][pd]
  __shared__ _Float16 Vs[2][PDc][LDK];    // [pd][key]
  __shared__ _Float16 Ps[8][16][KB];      // per-wave P staging (C->A relayout)

  const int tid  = threadIdx.x;
  const int lane = tid & 31;
  const int wave = tid >> 5;
  const int l16  = lane & 15;
  const int g    = lane >> 4;
  const int bh   = blockIdx.y;
  const int q0   = blockIdx.x * 128 + wave * 16;

  const _Float16* kbase = Kh + (size_t)bh * Sc * PDc;
  const _Float16* vbase = Vt + (size_t)bh * PDc * Sc;

  auto issue_kv = [&](int buf, int kb) {
#pragma unroll
    for (int it = 0; it < 2; ++it) {
      int chunk = tid + it * 256;         // 512 chunks per tile
      int r = chunk >> 3, c = (chunk & 7) * 8;
      async_copy_b128(&Ks[buf][r][c], kbase + (size_t)(kb + r) * PDc + c);
      async_copy_b128(&Vs[buf][r][c], vbase + (size_t)r * Sc + kb + c);
    }
  };

  // Q fragments (A layout), kept in registers for the whole pass.
  const _Float16* qp = Qh + ((size_t)bh * Sc + q0) * PDc;
  v16h aQ[2];
#pragma unroll
  for (int kk = 0; kk < 2; ++kk) {
    const _Float16* pr = qp + (size_t)l16 * PDc + kk * 32 + g * 8;
    aQ[kk] = ldfrag(pr, pr + 16);
  }

  v8f oacc[4];
  v8f zero = {};
#pragma unroll
  for (int ni = 0; ni < 4; ++ni) oacc[ni] = zero;
  float ml[8], ll[8];
#pragma unroll
  for (int i = 0; i < 8; ++i) { ml[i] = -1e30f; ll[i] = 0.0f; }

  issue_kv(0, 0);

  for (int kb = 0; kb < Sc; kb += KB) {
    const int buf = (kb / KB) & 1;
    wait_async0();
    __syncthreads();
    if (kb + KB < Sc) issue_kv(buf ^ 1, kb + KB);

    // scores S[16 x 64] = Q (16x64) @ K^T
    v8f sc[4];
#pragma unroll
    for (int ni = 0; ni < 4; ++ni) sc[ni] = zero;
#pragma unroll
    for (int kk = 0; kk < 2; ++kk)
#pragma unroll
      for (int ni = 0; ni < 4; ++ni) {
        const _Float16* pr = &Ks[buf][ni * 16 + l16][kk * 32 + g * 8];
        v16h bk = ldfrag(pr, pr + 16);
        sc[ni] = __builtin_amdgcn_wmma_f32_16x16x32_f16(
            false, aQ[kk], false, bk, (short)0, sc[ni], false, false);
      }

    // online softmax; row i + 8*g lives at element i across the 16-lane half
#pragma unroll
    for (int i = 0; i < 8; ++i) {
      float rmax = -1e30f;
#pragma unroll
      for (int ni = 0; ni < 4; ++ni) {
        float* sp = reinterpret_cast<float*>(&sc[ni]);
        sp[i] *= 0.125f;                       // 1/sqrt(PD)
        rmax = fmaxf(rmax, sp[i]);
      }
      rmax = fmaxf(rmax, __shfl_xor(rmax, 1, 32));
      rmax = fmaxf(rmax, __shfl_xor(rmax, 2, 32));
      rmax = fmaxf(rmax, __shfl_xor(rmax, 4, 32));
      rmax = fmaxf(rmax, __shfl_xor(rmax, 8, 32));
      float mnew  = fmaxf(ml[i], rmax);
      float alpha = __expf(ml[i] - mnew);
      ml[i] = mnew;
      float rsum = 0.0f;
#pragma unroll
      for (int ni = 0; ni < 4; ++ni) {
        float* sp = reinterpret_cast<float*>(&sc[ni]);
        float e = __expf(sp[i] - mnew);
        sp[i] = e;
        rsum += e;
      }
      rsum += __shfl_xor(rsum, 1, 32);
      rsum += __shfl_xor(rsum, 2, 32);
      rsum += __shfl_xor(rsum, 4, 32);
      rsum += __shfl_xor(rsum, 8, 32);
      ll[i] = ll[i] * alpha + rsum;
#pragma unroll
      for (int ni = 0; ni < 4; ++ni)
        reinterpret_cast<float*>(&oacc[ni])[i] *= alpha;
      // stage P into per-wave LDS (row = i+8g, col = ni*16+l16)
#pragma unroll
      for (int ni = 0; ni < 4; ++ni)
        Ps[wave][i + 8 * g][ni * 16 + l16] =
            (_Float16)reinterpret_cast<float*>(&sc[ni])[i];
    }
    // wave-local LDS RAW: DS ops are in-order per wave, but be explicit.
    asm volatile("s_wait_dscnt 0" ::: "memory");

    v16h aP[2];
#pragma unroll
    for (int kk = 0; kk < 2; ++kk) {
      const _Float16* pr = &Ps[wave][l16][kk * 32 + g * 8];
      aP[kk] = ldfrag(pr, pr + 16);
    }
#pragma unroll
    for (int kk = 0; kk < 2; ++kk)
#pragma unroll
      for (int ni = 0; ni < 4; ++ni) {
        const _Float16* pr = &Vs[buf][ni * 16 + l16][kk * 32 + g * 8];
        v16h bv = ldfrag(pr, pr + 16);
        oacc[ni] = __builtin_amdgcn_wmma_f32_16x16x32_f16(
            false, aP[kk], false, bv, (short)0, oacc[ni], false, false);
      }
  }

  // normalize and write attn output as f16 [B,S,D] row-major
  int b = bh >> 4, h = bh & 15;
#pragma unroll
  for (int ni = 0; ni < 4; ++ni) {
    const float* op = reinterpret_cast<const float*>(&oacc[ni]);
#pragma unroll
    for (int i = 0; i < 8; ++i) {
      int sq = q0 + i + 8 * g;
      float v = op[i] / ll[i];
      attn[((size_t)b * Sc + sq) * Dc + h * PDc + ni * 16 + l16] = (_Float16)v;
    }
  }
}

// ---------------------------------------------------------------------------
// Host-side launcher
// ---------------------------------------------------------------------------
extern "C" void kernel_launch(void* const* d_in, const int* in_sizes, int n_in,
                              void* d_out, int out_size, void* d_ws, size_t ws_size,
                              hipStream_t stream) {
  const float* x  = (const float*)d_in[0];
  const float* Wq = (const float*)d_in[1];
  const float* bq = (const float*)d_in[2];
  const float* Wk = (const float*)d_in[3];
  const float* bk = (const float*)d_in[4];
  const float* Wv = (const float*)d_in[5];
  const float* bv = (const float*)d_in[6];
  const float* Wo = (const float*)d_in[7];
  const float* bo = (const float*)d_in[8];

  _Float16* ws = (_Float16*)d_ws;
  size_t off = 0;
  _Float16* xh    = ws + off; off += (size_t)Mtot * Dc;   // 8M halves
  _Float16* WqT   = ws + off; off += (size_t)Dc * Dc;
  _Float16* WkT   = ws + off; off += (size_t)Dc * Dc;
  _Float16* WvT   = ws + off; off += (size_t)Dc * Dc;
  _Float16* WoT   = ws + off; off += (size_t)Dc * Dc;
  _Float16* Qh    = ws + off; off += (size_t)Bc * Hc * Sc * PDc;
  _Float16* Kh    = ws + off; off += (size_t)Bc * Hc * Sc * PDc;
  _Float16* Vth   = ws + off; off += (size_t)Bc * Hc * Sc * PDc;
  _Float16* attnh = ws + off; off += (size_t)Mtot * Dc;

  int nx4 = Mtot * Dc / 4;
  cvt_f32_f16<<<(nx4 + 255) / 256, 256, 0, stream>>>(x, xh, nx4);
  int nw = Dc * Dc;
  cvt_transpose<<<(nw + 255) / 256, 256, 0, stream>>>(Wq, WqT);
  cvt_transpose<<<(nw + 255) / 256, 256, 0, stream>>>(Wk, WkT);
  cvt_transpose<<<(nw + 255) / 256, 256, 0, stream>>>(Wv, WvT);
  cvt_transpose<<<(nw + 255) / 256, 256, 0, stream>>>(Wo, WoT);

  dim3 gg(Dc / 128, Mtot / 128);   // (8, 64)
  gemm_wmma<MODE_QK><<<gg, 256, 0, stream>>>(xh, WqT, bq, (void*)Qh);
  gemm_wmma<MODE_QK><<<gg, 256, 0, stream>>>(xh, WkT, bk, (void*)Kh);
  gemm_wmma<MODE_VT><<<gg, 256, 0, stream>>>(xh, WvT, bv, (void*)Vth);

  flash_attn<<<dim3(Sc / 128, Bc * Hc), 256, 0, stream>>>(Qh, Kh, Vth, attnh);

  gemm_wmma<MODE_F32><<<gg, 256, 0, stream>>>(attnh, WoT, bo, d_out);
}
